// RankOneMoE_74139725463721
// MI455X (gfx1250) — matrix operand order repaired and checked by reading, not compiled, not run on hardware
//
#include <hip/hip_runtime.h>

typedef __bf16 bf16_t;
typedef __attribute__((ext_vector_type(16))) __bf16 v16bf;
typedef __attribute__((ext_vector_type(8)))  float  v8f;

// Problem constants (B,S,H,F,E,K) = (16,512,768,3072,8,16)
static constexpr int  Bb   = 16;
static constexpr int  Ss   = 512;
static constexpr int  Hh   = 768;
static constexpr int  Ff   = 3072;
static constexpr int  Ee   = 8;
static constexpr int  EK   = 128;        // E*K
static constexpr int  BS   = Bb * Ss;    // 8192
static constexpr int  G1N  = 2 * EK;     // 256
static constexpr int  KC1  = Hh + EK;    // 896  (Xcat width)
static constexpr int  KC2  = Ff + EK;    // 3200 (Hcat width)
static constexpr float LAMBDA = 0.2f;

__device__ __forceinline__ v16bf frag_pack(uint4 lo, uint4 hi) {
    union { uint4 u[2]; v16bf v; } t;
    t.u[0] = lo; t.u[1] = hi;
    return t.v;
}

// Async 16-byte global -> LDS copy (CDNA5, tracked with ASYNCcnt).
// VDST operand carries the workgroup-relative LDS byte offset, which is the
// low 32 bits of the generic pointer to a __shared__ object.
__device__ __forceinline__ void async_copy16(bf16_t* lds_dst, const bf16_t* gsrc) {
    unsigned ldsoff = (unsigned)(unsigned long long)lds_dst;
    unsigned long long ga = (unsigned long long)gsrc;
    asm volatile("global_load_async_to_lds_b128 %0, %1, off"
                 :: "v"(ldsoff), "v"(ga) : "memory");
}

// ---------------------------------------------------------------------------
// Generic WMMA GEMM:  C[m,n] = sum_k A[m,k] * B[n,k]  (+bias[n]) (opt relu)
// A: bf16 row-major [M, lda]; B: bf16 [N, ldb] (N-major, i.e. already B^T)
// If SCALE: multiply the raw accumulator by gw[(m>>9)*128 + n] before bias.
// Block: 256 threads (8 waves), tile 256x128, BK=32, grid (N/128, M/256).
// 8 waves arranged 4(M) x 2(N); each wave owns 64x64 = 16 WMMA tiles.
// Double-buffered LDS fed by global_load_async_to_lds_b128.
// ---------------------------------------------------------------------------
template<bool OUT_BF16, bool RELU, bool SCALE>
__global__ __launch_bounds__(256)
void wmma_gemm_bf16(const bf16_t* __restrict__ A, int lda,
                    const bf16_t* __restrict__ Bm, int ldb,
                    void* __restrict__ Cout, int ldc, int K,
                    const float* __restrict__ bias,
                    const float* __restrict__ gw)
{
    __shared__ bf16_t As[2][256 * 40];   // 32 cols padded to 40 (conflict-free)
    __shared__ bf16_t Bs[2][128 * 40];

    const int tid   = threadIdx.x;
    const int lane  = tid & 31;
    const int wave  = tid >> 5;
    const int wm    = wave >> 1;      // 0..3  (64-row slab)
    const int wn    = wave & 1;       // 0..1  (64-col slab)
    const int lrow  = lane & 15;
    const int lhalf = lane >> 4;

    const long blockM = (long)blockIdx.y * 256;
    const long blockN = (long)blockIdx.x * 128;

    v8f acc[4][4];
    #pragma unroll
    for (int i = 0; i < 4; ++i)
        #pragma unroll
        for (int j = 0; j < 4; ++j) {
            v8f z = {0.f, 0.f, 0.f, 0.f, 0.f, 0.f, 0.f, 0.f};
            acc[i][j] = z;
        }

    // Stage one 256x32 A tile + 128x32 B tile into LDS buffer `p` (async).
    // 6 async-load instructions per wave per tile.
    auto stage = [&](int k0, int p) {
        #pragma unroll
        for (int i = 0; i < 4; ++i) {               // A: 1024 chunks of 8 bf16
            const int c  = tid + i * 256;
            const int r  = c >> 2;
            const int kc = (c & 3) * 8;
            async_copy16(&As[p][r * 40 + kc],
                         A + (blockM + r) * (long)lda + k0 + kc);
        }
        #pragma unroll
        for (int i = 0; i < 2; ++i) {               // B: 512 chunks of 8 bf16
            const int c  = tid + i * 256;
            const int r  = c >> 2;
            const int kc = (c & 3) * 8;
            async_copy16(&Bs[p][r * 40 + kc],
                         Bm + (blockN + r) * (long)ldb + k0 + kc);
        }
    };

    stage(0, 0);
    int buf = 0;
    for (int k0 = 0; k0 < K; k0 += 32) {
        if (k0 + 32 < K) {
            stage(k0 + 32, buf ^ 1);                       // prefetch next tile
            asm volatile("s_wait_asynccnt 0x6" ::: "memory"); // current tile done
        } else {
            asm volatile("s_wait_asynccnt 0x0" ::: "memory");
        }
        __syncthreads();

        // --- fragments (ISA 7.12.2 16-bit A 16x32 / B 32x16 layouts) ----
        v16bf afrag[4];
        #pragma unroll
        for (int mt = 0; mt < 4; ++mt) {
            const int row = wm * 64 + mt * 16 + lrow;
            const int kc  = lhalf * 8;                          // 0 or 8
            uint4 lo = *(const uint4*)(&As[buf][row * 40 + kc]);      // K=kc..kc+7
            uint4 hi = *(const uint4*)(&As[buf][row * 40 + kc + 16]); // K=kc+16..+23
            afrag[mt] = frag_pack(lo, hi);
        }
        v16bf bfrag[4];
        #pragma unroll
        for (int nt = 0; nt < 4; ++nt) {
            const int n  = wn * 64 + nt * 16 + lrow;
            const int kb = lhalf * 8;                           // 0 or 8
            uint4 lo = *(const uint4*)(&Bs[buf][n * 40 + kb]);        // K=kb..kb+7
            uint4 hi = *(const uint4*)(&Bs[buf][n * 40 + kb + 16]);   // K=kb+16..+23
            bfrag[nt] = frag_pack(lo, hi);
        }

        #pragma unroll
        for (int mt = 0; mt < 4; ++mt)
            #pragma unroll
            for (int nt = 0; nt < 4; ++nt)
                acc[mt][nt] = __builtin_amdgcn_wmma_f32_16x16x32_bf16(
                    false, afrag[mt], false, bfrag[nt],
                    (short)0, acc[mt][nt], false, false);
        __syncthreads();
        buf ^= 1;
    }

    // --- epilogue: scale / bias / relu / convert / store --------------------
    #pragma unroll
    for (int mt = 0; mt < 4; ++mt) {
        #pragma unroll
        for (int nt = 0; nt < 4; ++nt) {
            const long n = blockN + wn * 64 + nt * 16 + lrow;
            float sc = 1.f;
            if constexpr (SCALE) sc = gw[(blockM >> 9) * 128 + n];
            const float bv = bias ? bias[n] : 0.f;
            #pragma unroll
            for (int r = 0; r < 8; ++r) {
                const long m = blockM + wm * 64 + mt * 16 + r + lhalf * 8;
                float v = acc[mt][nt][r];
                if constexpr (SCALE) v *= sc;
                v += bv;
                if constexpr (RELU) v = fmaxf(v, 0.f);
                if constexpr (OUT_BF16)
                    ((bf16_t*)Cout)[m * (long)ldc + n] = (bf16_t)v;
                else
                    ((float*)Cout)[m * (long)ldc + n] = v;
            }
        }
    }
}

// ------------------------- small helper kernels ----------------------------
// fp32 [R, ldsrc] -> bf16 [R, lddst] (convert)
__global__ void cvt2d(const float* __restrict__ src, int ldsrc,
                      bf16_t* __restrict__ dst, int lddst, int C)
{
    const int r = blockIdx.y;
    const int c = blockIdx.x * 256 + threadIdx.x;
    if (c < C) dst[(long)r * lddst + c] = (bf16_t)src[(long)r * ldsrc + c];
}

// fp32 [R, C] -> bf16 transposed into dst[c*lddst + coloff + r]
__global__ void cvtT(const float* __restrict__ src, int C,
                     bf16_t* __restrict__ dst, int lddst, int coloff)
{
    const int r = blockIdx.y;
    const int c = blockIdx.x * 256 + threadIdx.x;
    if (c < C) dst[(long)c * lddst + coloff + r] = (bf16_t)src[(long)r * C + c];
}

// out[i] = b[i] + 0.2 * sum_e TB[e, i]
__global__ void bias_merge(const float* __restrict__ b,
                           const float* __restrict__ TB,
                           int N, float* __restrict__ out)
{
    const int i = blockIdx.x * 256 + threadIdx.x;
    if (i < N) {
        float s = b[i];
        #pragma unroll
        for (int e = 0; e < Ee; ++e) s += LAMBDA * TB[(long)e * N + i];
        out[i] = s;
    }
}

// gw[b, g] = mean_s G2[b, s, g]
__global__ void gw_mean(const float* __restrict__ G2, float* __restrict__ gw)
{
    const int b = blockIdx.x;      // 16 blocks
    const int g = threadIdx.x;     // 128 threads
    const float* p = G2 + (long)b * Ss * EK + g;
    float s = 0.f;
    for (int i = 0; i < Ss; ++i) s += p[(long)i * EK];
    gw[b * EK + g] = s * (1.f / (float)Ss);
}

// ---------------------------------------------------------------------------
extern "C" void kernel_launch(void* const* d_in, const int* in_sizes, int n_in,
                              void* d_out, int out_size, void* d_ws, size_t ws_size,
                              hipStream_t stream)
{
    (void)in_sizes; (void)n_in; (void)out_size; (void)ws_size;
    const float* x        = (const float*)d_in[0];
    const float* gate_w1  = (const float*)d_in[1];
    const float* gate_b1  = (const float*)d_in[2];
    const float* gate_w2  = (const float*)d_in[3];
    const float* gate_b2  = (const float*)d_in[4];
    const float* W1       = (const float*)d_in[5];
    const float* b1       = (const float*)d_in[6];
    const float* W2       = (const float*)d_in[7];
    const float* b2       = (const float*)d_in[8];
    const float* U1       = (const float*)d_in[9];
    const float* SVH1     = (const float*)d_in[10];
    const float* U2       = (const float*)d_in[11];
    const float* SVH2     = (const float*)d_in[12];
    const float* TB1      = (const float*)d_in[13];
    const float* TB2      = (const float*)d_in[14];

    // workspace carve-up (all offsets 256B aligned)
    char* ws = (char*)d_ws;
    size_t off = 0;
    auto alloc = [&](size_t bytes) {
        void* p = ws + off;
        off += (bytes + 255) & ~(size_t)255;
        return p;
    };
    bf16_t* Xcat  = (bf16_t*)alloc((size_t)BS * KC1 * 2);   // [8192, 896]
    bf16_t* Wg1   = (bf16_t*)alloc((size_t)G1N * Hh * 2);   // [256, 768]
    bf16_t* Wg2   = (bf16_t*)alloc((size_t)EK * G1N * 2);   // [128, 256]
    bf16_t* G1    = (bf16_t*)alloc((size_t)BS * G1N * 2);   // [8192, 256]
    float*  G2f   = (float*) alloc((size_t)BS * EK * 4);    // [8192, 128]
    float*  gw    = (float*) alloc((size_t)Bb * EK * 4);    // [16, 128]
    bf16_t* SVH1b = (bf16_t*)alloc((size_t)EK * Hh * 2);    // [128, 768]
    bf16_t* Wcat1 = (bf16_t*)alloc((size_t)Ff * KC1 * 2);   // [3072, 896]
    float*  b1m   = (float*) alloc((size_t)Ff * 4);
    bf16_t* Hcat  = (bf16_t*)alloc((size_t)BS * KC2 * 2);   // [8192, 3200]
    bf16_t* SVH2b = (bf16_t*)alloc((size_t)EK * Ff * 2);    // [128, 3072]
    bf16_t* Wcat2 = (bf16_t*)alloc((size_t)Hh * KC2 * 2);   // [768, 3200]
    float*  b2m   = (float*) alloc((size_t)Hh * 4);

    // ---- conversions to bf16 (and transposed embedding of U1/U2) ----------
    cvt2d<<<dim3(3, BS),   256, 0, stream>>>(x,       Hh,  Xcat,  KC1, Hh);
    cvt2d<<<dim3(3, G1N),  256, 0, stream>>>(gate_w1, Hh,  Wg1,   Hh,  Hh);
    cvt2d<<<dim3(1, EK),   256, 0, stream>>>(gate_w2, G1N, Wg2,   G1N, G1N);
    cvt2d<<<dim3(3, Ff),   256, 0, stream>>>(W1,      Hh,  Wcat1, KC1, Hh);
    cvtT <<<dim3(12, EK),  256, 0, stream>>>(U1,      Ff,  Wcat1, KC1, Hh);
    cvt2d<<<dim3(3, EK),   256, 0, stream>>>(SVH1,    Hh,  SVH1b, Hh,  Hh);
    cvt2d<<<dim3(12, Hh),  256, 0, stream>>>(W2,      Ff,  Wcat2, KC2, Ff);
    cvtT <<<dim3(3, EK),   256, 0, stream>>>(U2,      Hh,  Wcat2, KC2, Ff);
    cvt2d<<<dim3(12, EK),  256, 0, stream>>>(SVH2,    Ff,  SVH2b, Ff,  Ff);
    bias_merge<<<dim3(12), 256, 0, stream>>>(b1, TB1, Ff, b1m);
    bias_merge<<<dim3(3),  256, 0, stream>>>(b2, TB2, Hh, b2m);

    const dim3 blk(256);
    const int MB = BS / 256;   // 32 row-blocks

    // ---- router -----------------------------------------------------------
    // G1 = relu(x @ gate_w1^T + gate_b1)            [8192, 256] bf16
    wmma_gemm_bf16<true, true, false><<<dim3(G1N / 128, MB), blk, 0, stream>>>(
        Xcat, KC1, Wg1, Hh, G1, G1N, Hh, gate_b1, nullptr);
    // G2 = G1 @ gate_w2^T + gate_b2                 [8192, 128] fp32
    wmma_gemm_bf16<false, false, false><<<dim3(EK / 128, MB), blk, 0, stream>>>(
        G1, G1N, Wg2, G1N, G2f, EK, G1N, gate_b2, nullptr);
    // gw = mean over S
    gw_mean<<<dim3(Bb), dim3(EK), 0, stream>>>(G2f, gw);

    // ---- fc1 --------------------------------------------------------------
    // T1 = (x @ SVH1^T) * gw  -> Xcat[:, 768:896]   bf16
    wmma_gemm_bf16<true, false, true><<<dim3(EK / 128, MB), blk, 0, stream>>>(
        Xcat, KC1, SVH1b, Hh, Xcat + Hh, KC1, Hh, nullptr, gw);
    // H = relu(Xcat @ Wcat1^T + b1m) -> Hcat[:, :3072]  bf16
    wmma_gemm_bf16<true, true, false><<<dim3(Ff / 128, MB), blk, 0, stream>>>(
        Xcat, KC1, Wcat1, KC1, Hcat, KC2, KC1, b1m, nullptr);

    // ---- fc2 --------------------------------------------------------------
    // T2 = (H @ SVH2^T) * gw -> Hcat[:, 3072:3200]  bf16
    wmma_gemm_bf16<true, false, true><<<dim3(EK / 128, MB), blk, 0, stream>>>(
        Hcat, KC2, SVH2b, Ff, Hcat + Ff, KC2, Ff, nullptr, gw);
    // out = Hcat @ Wcat2^T + b2m                    [8192, 768] fp32
    wmma_gemm_bf16<false, false, false><<<dim3(Hh / 128, MB), blk, 0, stream>>>(
        Hcat, KC2, Wcat2, KC2, (float*)d_out, Hh, KC2, b2m, nullptr);
}
